// FusedMoEExperts_8186207666933
// MI455X (gfx1250) — matrix operand order repaired and checked by reading, not compiled, not run on hardware
//
#include <hip/hip_runtime.h>
#include <hip/hip_bf16.h>

// ---------------- problem constants (from reference setup_inputs) ----------------
#define E_    16
#define D_    1024
#define ED_   512
#define NB_   8
#define DB_   64
#define R_    2
#define F_    256
#define C_    2048
#define N_    (E_ * C_)
#define TM    16            // tokens per block tile
#define TILES (C_ / TM)     // 128 tiles per expert

typedef __attribute__((ext_vector_type(16))) __bf16 v16bf;
typedef __attribute__((ext_vector_type(8)))  float  v8f;

// ---------------- Cl(3,0) Cayley pair table ----------------
// For each output blade k: 8 (i, j, sign) pairs with blade_i * blade_j = sign * blade_k.
// Blade order: 1, e1, e2, e3, e12, e13, e23, e123.
__constant__ const int CI[8][8] = {
  {0,1,2,3,4,5,6,7}, {0,1,2,4,3,5,6,7}, {0,2,1,4,3,6,5,7}, {0,3,1,5,2,6,4,7},
  {0,4,1,2,3,7,5,6}, {0,5,1,3,2,7,4,6}, {0,6,2,3,1,7,4,5}, {0,7,1,6,2,5,3,4}};
__constant__ const int CJ[8][8] = {
  {0,1,2,3,4,5,6,7}, {1,0,4,2,5,3,7,6}, {2,0,4,1,6,3,7,5}, {3,0,5,1,6,2,7,4},
  {4,0,2,1,7,3,6,5}, {5,0,3,1,7,2,6,4}, {6,0,3,2,7,1,5,4}, {7,0,6,1,5,2,4,3}};
__constant__ const float CS[8][8] = {
  { 1, 1, 1, 1,-1,-1,-1,-1}, { 1, 1,-1, 1,-1, 1,-1,-1},
  { 1, 1, 1,-1,-1, 1, 1, 1}, { 1, 1, 1,-1, 1,-1,-1,-1},
  { 1, 1, 1,-1, 1, 1,-1, 1}, { 1, 1, 1,-1,-1,-1, 1,-1},
  { 1, 1, 1,-1, 1, 1,-1, 1}, { 1, 1, 1, 1,-1,-1, 1, 1}};

// ---------------- small helpers ----------------
__device__ __forceinline__ __bf16 f2bf(float f) { return (__bf16)f; }  // v_cvt_bf16_f32 (RNE)
__device__ __forceinline__ float bf2f(__bf16 b) {
  unsigned short h; __builtin_memcpy(&h, &b, 2);
  return __uint_as_float(((unsigned)h) << 16);
}
__device__ __forceinline__ float sigm(float x) { return 1.f / (1.f + __expf(-x)); }

__device__ __forceinline__ v8f wmma_bf16(v16bf a, v16bf b, v8f c) {
  return __builtin_amdgcn_wmma_f32_16x16x32_bf16(false, a, false, b, (short)0, c, false, false);
}

// A fragment (16x32 bf16): lane holds row M=lane&15; K pairs per ISA layout table.
// p must already point at row_base + k0 + (lane>>4)*8 ; f32 source.
__device__ __forceinline__ v16bf lda_f32(const float* p) {
  v16bf a;
#pragma unroll
  for (int v = 0; v < 8; ++v) {
    const int kb = ((v & 4) ? 16 : 0) + 2 * (v & 3);
    a[2 * v]     = f2bf(p[kb]);
    a[2 * v + 1] = f2bf(p[kb + 1]);
  }
  return a;
}
// Same, bf16 source (LDS flat/h tiles or pre-converted global x).
__device__ __forceinline__ v16bf lda_bf(const __bf16* p) {
  v16bf a;
#pragma unroll
  for (int v = 0; v < 8; ++v) {
    const int kb = ((v & 4) ? 16 : 0) + 2 * (v & 3);
    a[2 * v]     = p[kb];
    a[2 * v + 1] = p[kb + 1];
  }
  return a;
}
// B fragment (32x16): lane holds column N = n0 + (lane&15); K pairs per ISA layout table.
// p must already point at (k0 + (lane>>4)*8) * ld + n0 + (lane&15).
__device__ __forceinline__ v16bf ldb_f32(const float* p, int ld) {
  v16bf b;
#pragma unroll
  for (int v = 0; v < 8; ++v) {
    const int kb = ((v & 4) ? 16 : 0) + 2 * (v & 3);
    b[2 * v]     = f2bf(p[(size_t)kb * ld]);
    b[2 * v + 1] = f2bf(p[(size_t)(kb + 1) * ld]);
  }
  return b;
}
// bf16 B fragment: same strided column gather.
__device__ __forceinline__ v16bf ldb_bf16(const __bf16* p, int ld) {
  v16bf b;
#pragma unroll
  for (int v = 0; v < 8; ++v) {
    const int kb = ((v & 4) ? 16 : 0) + 2 * (v & 3);
    b[2 * v]     = p[(size_t)kb * ld];
    b[2 * v + 1] = p[(size_t)(kb + 1) * ld];
  }
  return b;
}

template<int MODE>  // MODE>=1: bf16 weights in ws; MODE>=2: bf16 x in ws too
__device__ __forceinline__ v16bf ldB(const float* pf, const __bf16* pb, int ld) {
  if constexpr (MODE >= 1) return ldb_bf16(pb, ld); else return ldb_f32(pf, ld);
}

// ---------------- f32 -> bf16 weight/activation prep ----------------
__global__ __launch_bounds__(256) void cvt_f32_bf16_kernel(
    const float* __restrict__ s, __bf16* __restrict__ d, int n4) {
  int i = blockIdx.x * blockDim.x + threadIdx.x;
  const int stride = gridDim.x * blockDim.x;
  for (; i < n4; i += stride) {
    const float4 v = ((const float4*)s)[i];
    __bf16 o[4] = { f2bf(v.x), f2bf(v.y), f2bf(v.z), f2bf(v.w) };
    __builtin_memcpy(d + (size_t)i * 4, o, 8);
  }
}

// ---------------- fused MoE expert kernel ----------------
template<int MODE>
__global__ __launch_bounds__(256) void fused_moe_experts_kernel(
    const float* __restrict__ x,       // (N, D) f32
    const __bf16* __restrict__ xb,     // (N, D) bf16 (MODE>=2)
    const float* __restrict__ p_in_f,  const __bf16* __restrict__ p_in_b,   // (E, D, ED)
    const float* __restrict__ p_out_f, const __bf16* __restrict__ p_out_b,  // (E, ED, D)
    const float* __restrict__ wg_f,    const __bf16* __restrict__ wg_b,     // (E, R, DB, F)
    const float* __restrict__ wu_f,    const __bf16* __restrict__ wu_b,     // (E, R, DB, F)
    const float* __restrict__ wd_f,    const __bf16* __restrict__ wd_b,     // (E, R, F, DB)
    const float* __restrict__ iw,      // (E, R, 64)
    const float* __restrict__ gg,      // (E, R)
    const float* __restrict__ lnw,     // (E, R, DB)
    const float* __restrict__ lnb,     // (E, R, DB)
    const float* __restrict__ olnw,    // (D,)
    const float* __restrict__ olnb,    // (D,)
    const int*   __restrict__ eoffs,   // (E,)
    float* __restrict__ out)           // (N, D)
{
  __shared__ float  mvS[TM][ED_];            // 32 KB, persistent multivector tile
  __shared__ __bf16 flatS[TM * NB_][DB_];    // 16 KB, post-LN FFN input
  __shared__ __bf16 hS[8][16][32];           // 8 KB, per-wave h bounce (D-frag -> A-frag)
  __shared__ float  wS[64], lnwS[DB_], lnbS[DB_];
  __shared__ float  statSum[TM][NB_], statSq[TM][NB_];
  __shared__ float  oSum[TM], oSq[TM];

  const int tid  = threadIdx.x;
  const int lane = tid & 31;
  const int wv   = tid >> 5;        // wave id 0..7
  const int half = lane >> 4;
  const int l15  = lane & 15;
  const int e    = blockIdx.x / TILES;
  const int tile = blockIdx.x % TILES;
  const int row0 = eoffs[e] + tile * TM;

  // ================= Phase 1: mv = x_tile (16x1024) @ p_in[e] (1024x512) =================
  {
    const float*  Bpf  = p_in_f + (size_t)e * D_ * ED_;
    const __bf16* Bpb  = p_in_b + (size_t)e * D_ * ED_;
    const float*  Arf  = x  + (size_t)(row0 + l15) * D_ + half * 8;
    const __bf16* Arb  = xb + (size_t)(row0 + l15) * D_ + half * 8;
#pragma unroll 1
    for (int nt = 0; nt < 4; ++nt) {
      const int n0 = wv * 64 + nt * 16;
      v8f acc = {};
#pragma unroll 1
      for (int kc = 0; kc < 32; ++kc) {
        v16bf a;
        if constexpr (MODE >= 2) a = lda_bf(Arb + kc * 32);
        else                     a = lda_f32(Arf + kc * 32);
        const v16bf b = ldB<MODE>(Bpf + (size_t)(kc * 32 + half * 8) * ED_ + n0 + l15,
                                  Bpb + (size_t)(kc * 32 + half * 8) * ED_ + n0 + l15, ED_);
        acc = wmma_bf16(a, b, acc);
      }
#pragma unroll
      for (int v = 0; v < 8; ++v) mvS[v + 8 * half][n0 + l15] = acc[v];
    }
  }
  __syncthreads();

  // ================= Phase 2: two geometric-FFN rounds (mv updated in LDS) =================
#pragma unroll 1
  for (int r = 0; r < R_; ++r) {
    const int pb = (e * R_ + r) * 64;
    if (tid < 64) {
      wS[tid]   = sigm(iw[pb + tid]);
      lnwS[tid] = lnw[pb + tid];
      lnbS[tid] = lnb[pb + tid];
    }
    if (tid < TM * NB_) { statSum[tid >> 3][tid & 7] = 0.f; statSq[tid >> 3][tid & 7] = 0.f; }
    __syncthreads();

    const float gate = sigm(gg[e * R_ + r]);
    float wloc[64];
#pragma unroll
    for (int q = 0; q < 64; ++q) wloc[q] = wS[q];

    // geo -> gate-mix -> LN stats; 1024 (token,d) columns, 4 per thread
#pragma unroll 1
    for (int c = 0; c < 4; ++c) {
      const int col = c * 256 + tid;
      const int n = col >> 6, d = col & 63;
      float m[8];
#pragma unroll
      for (int i = 0; i < 8; ++i) m[i] = mvS[n][i * 64 + d];
#pragma unroll
      for (int k = 0; k < 8; ++k) {
        float g = 0.f;
#pragma unroll
        for (int q = 0; q < 8; ++q) {
          const int i = CI[k][q], j = CJ[k][q];
          g += CS[k][q] * wloc[i * 8 + j] * m[i] * m[j];
        }
        const float mx = gate * g + (1.f - gate) * m[k];
        flatS[n * 8 + k][d] = f2bf(mx);
        atomicAdd(&statSum[n][k], mx);
        atomicAdd(&statSq[n][k], mx * mx);
      }
    }
    __syncthreads();

    // LayerNorm over DB=64 in place (bf16)
#pragma unroll 1
    for (int c = 0; c < 4; ++c) {
      const int col = c * 256 + tid;
      const int n = col >> 6, d = col & 63;
#pragma unroll
      for (int k = 0; k < 8; ++k) {
        const float mu  = statSum[n][k] * (1.f / 64.f);
        const float var = statSq[n][k] * (1.f / 64.f) - mu * mu;
        const float rs  = rsqrtf(var + 1e-5f);
        const float v   = bf2f(flatS[n * 8 + k][d]);
        flatS[n * 8 + k][d] = f2bf((v - mu) * rs * lnwS[d] + lnbS[d]);
      }
    }
    __syncthreads();

    // SwiGLU FFN: each wave owns 16 flat rows (M-tile). K=64, F=256, down back to 64.
    const size_t wgo = (size_t)(e * R_ + r) * DB_ * F_;
    const size_t wdo = (size_t)(e * R_ + r) * F_ * DB_;
    const float*  Wgf = wg_f + wgo;  const __bf16* Wgb = wg_b + wgo;
    const float*  Wuf = wu_f + wgo;  const __bf16* Wub = wu_b + wgo;
    const float*  Wdf = wd_f + wdo;  const __bf16* Wdb = wd_b + wdo;
    const __bf16* fA = &flatS[16 * wv + l15][0] + half * 8;
    const v16bf a0 = lda_bf(fA);       // K = 0..31
    const v16bf a1 = lda_bf(fA + 32);  // K = 32..63
    v8f dacc[4];
    {
      v8f z = {};
#pragma unroll
      for (int nt = 0; nt < 4; ++nt) dacc[nt] = z;
    }
#pragma unroll 1
    for (int ft = 0; ft < 8; ++ft) {   // F in chunks of 32 (= down-GEMM K chunk)
#pragma unroll
      for (int sub = 0; sub < 2; ++sub) {
        const int n0 = ft * 32 + sub * 16;
        v8f ga = {}, ua = {};
        ga = wmma_bf16(a0, ldB<MODE>(Wgf + (size_t)(half * 8)      * F_ + n0 + l15,
                                     Wgb + (size_t)(half * 8)      * F_ + n0 + l15, F_), ga);
        ga = wmma_bf16(a1, ldB<MODE>(Wgf + (size_t)(32 + half * 8) * F_ + n0 + l15,
                                     Wgb + (size_t)(32 + half * 8) * F_ + n0 + l15, F_), ga);
        ua = wmma_bf16(a0, ldB<MODE>(Wuf + (size_t)(half * 8)      * F_ + n0 + l15,
                                     Wub + (size_t)(half * 8)      * F_ + n0 + l15, F_), ua);
        ua = wmma_bf16(a1, ldB<MODE>(Wuf + (size_t)(32 + half * 8) * F_ + n0 + l15,
                                     Wub + (size_t)(32 + half * 8) * F_ + n0 + l15, F_), ua);
#pragma unroll
        for (int v = 0; v < 8; ++v) {
          const float g = ga[v];
          const float h = (g * sigm(g)) * ua[v];     // silu(gate) * up
          hS[wv][v + 8 * half][sub * 16 + l15] = f2bf(h);
        }
      }
      __syncthreads();                                // hS visible across lanes
      const v16bf ah = lda_bf(&hS[wv][l15][0] + half * 8);
#pragma unroll
      for (int nt = 0; nt < 4; ++nt)
        dacc[nt] = wmma_bf16(ah, ldB<MODE>(Wdf + (size_t)(ft * 32 + half * 8) * DB_ + nt * 16 + l15,
                                           Wdb + (size_t)(ft * 32 + half * 8) * DB_ + nt * 16 + l15, DB_),
                             dacc[nt]);
      __syncthreads();                                // protect hS before next chunk
    }
    // residual: mv += h @ w_down
    {
      const int rowb = 16 * wv;
#pragma unroll
      for (int v = 0; v < 8; ++v) {
        const int rr = rowb + v + 8 * half;
        const int n = rr >> 3, k = rr & 7;
#pragma unroll
        for (int nt = 0; nt < 4; ++nt)
          mvS[n][k * 64 + nt * 16 + l15] += dacc[nt][v];
      }
    }
    __syncthreads();
  }

  // ================= Phase 3: out = LN( mv (16x512) @ p_out[e] (512x1024) ) =================
  if (tid < TM) { oSum[tid] = 0.f; oSq[tid] = 0.f; }
  __syncthreads();
  {
    const float*  Bof = p_out_f + (size_t)e * ED_ * D_;
    const __bf16* Bob = p_out_b + (size_t)e * ED_ * D_;
    v8f oacc[8];
    {
      v8f z = {};
#pragma unroll
      for (int nt = 0; nt < 8; ++nt) oacc[nt] = z;
    }
#pragma unroll 1
    for (int kc = 0; kc < 16; ++kc) {
      const v16bf a = lda_f32(&mvS[l15][0] + kc * 32 + half * 8);
#pragma unroll
      for (int nt = 0; nt < 8; ++nt) {
        const size_t boff = (size_t)(kc * 32 + half * 8) * D_ + wv * 128 + nt * 16 + l15;
        oacc[nt] = wmma_bf16(a, ldB<MODE>(Bof + boff, Bob + boff, D_), oacc[nt]);
      }
    }
    // per-row stats for LN over D=1024 (LDS float atomics)
#pragma unroll
    for (int v = 0; v < 8; ++v) {
      float s = 0.f, sq = 0.f;
#pragma unroll
      for (int nt = 0; nt < 8; ++nt) { const float xv = oacc[nt][v]; s += xv; sq += xv * xv; }
      const int m = v + 8 * half;
      atomicAdd(&oSum[m], s);
      atomicAdd(&oSq[m], sq);
    }
    __syncthreads();
#pragma unroll
    for (int v = 0; v < 8; ++v) {
      const int m = v + 8 * half;
      const float mu  = oSum[m] * (1.f / (float)D_);
      const float var = oSq[m] * (1.f / (float)D_) - mu * mu;
      const float rs  = rsqrtf(var + 1e-5f);
      float* orow = out + (size_t)(row0 + m) * D_;
#pragma unroll
      for (int nt = 0; nt < 8; ++nt) {
        const int cc = wv * 128 + nt * 16 + l15;
        orow[cc] = (oacc[nt][v] - mu) * rs * olnw[cc] + olnb[cc];
      }
    }
  }
}

// ---------------- host launcher ----------------
extern "C" void kernel_launch(void* const* d_in, const int* in_sizes, int n_in,
                              void* d_out, int out_size, void* d_ws, size_t ws_size,
                              hipStream_t stream) {
  (void)in_sizes; (void)n_in; (void)out_size;
  const float* x     = (const float*)d_in[0];
  const float* p_in  = (const float*)d_in[1];
  const float* p_out = (const float*)d_in[2];
  const float* iw    = (const float*)d_in[3];
  const float* gg    = (const float*)d_in[4];
  const float* wg    = (const float*)d_in[5];
  const float* wu    = (const float*)d_in[6];
  const float* wd    = (const float*)d_in[7];
  const float* lnw   = (const float*)d_in[8];
  const float* lnb   = (const float*)d_in[9];
  const float* olnw  = (const float*)d_in[10];
  const float* olnb  = (const float*)d_in[11];
  const int*   eoffs = (const int*)d_in[12];
  float* outp = (float*)d_out;

  // bf16 staging plan in workspace
  const size_t n_pin  = (size_t)E_ * D_ * ED_;      // 8,388,608
  const size_t n_pout = (size_t)E_ * ED_ * D_;      // 8,388,608
  const size_t n_w    = (size_t)E_ * R_ * DB_ * F_; //   524,288 (each of wg/wu/wd)
  const size_t n_x    = (size_t)N_ * D_;            // 33,554,432
  const size_t w_elems = n_pin + n_pout + 3 * n_w;
  const bool bfw = ws_size >= 2 * w_elems;            // weights fit as bf16
  const bool bfx = ws_size >= 2 * (w_elems + n_x);    // x also fits

  __bf16* wsb = (__bf16*)d_ws;
  __bf16* pinb  = wsb;
  __bf16* poutb = pinb + n_pin;
  __bf16* wgb   = poutb + n_pout;
  __bf16* wub   = wgb + n_w;
  __bf16* wdb   = wub + n_w;
  __bf16* xbb   = wdb + n_w;

  if (bfw) {
    cvt_f32_bf16_kernel<<<2048, 256, 0, stream>>>(p_in,  pinb,  (int)(n_pin  / 4));
    cvt_f32_bf16_kernel<<<2048, 256, 0, stream>>>(p_out, poutb, (int)(n_pout / 4));
    cvt_f32_bf16_kernel<<<512,  256, 0, stream>>>(wg,    wgb,   (int)(n_w / 4));
    cvt_f32_bf16_kernel<<<512,  256, 0, stream>>>(wu,    wub,   (int)(n_w / 4));
    cvt_f32_bf16_kernel<<<512,  256, 0, stream>>>(wd,    wdb,   (int)(n_w / 4));
  }
  if (bfx) {
    cvt_f32_bf16_kernel<<<4096, 256, 0, stream>>>(x, xbb, (int)(n_x / 4));
  }

  // keep bf16 pointers valid even when unused (avoid null arithmetic)
  if (!bfw) { pinb = poutb = wgb = wub = wdb = xbb = (__bf16*)d_in[0]; }
  if (!bfx) { xbb = pinb; }

  const dim3 grid(E_ * TILES), blk(256);
  if (bfx) {
    fused_moe_experts_kernel<2><<<grid, blk, 0, stream>>>(
        x, xbb, p_in, pinb, p_out, poutb, wg, wgb, wu, wub, wd, wdb,
        iw, gg, lnw, lnb, olnw, olnb, eoffs, outp);
  } else if (bfw) {
    fused_moe_experts_kernel<1><<<grid, blk, 0, stream>>>(
        x, xbb, p_in, pinb, p_out, poutb, wg, wgb, wu, wub, wd, wdb,
        iw, gg, lnw, lnb, olnw, olnb, eoffs, outp);
  } else {
    fused_moe_experts_kernel<0><<<grid, blk, 0, stream>>>(
        x, xbb, p_in, pinb, p_out, poutb, wg, wgb, wu, wub, wd, wdb,
        iw, gg, lnw, lnb, olnw, olnb, eoffs, outp);
  }
}
